// CausalSelfAttention_83270825935200
// MI455X (gfx1250) — matrix operand order repaired and checked by reading, not compiled, not run on hardware
//
#include <hip/hip_runtime.h>

typedef unsigned short u16;
typedef unsigned int   u32;
typedef unsigned long long u64;

typedef float  v8f   __attribute__((ext_vector_type(8)));
typedef __bf16 v16bf __attribute__((ext_vector_type(16)));
typedef u32    u32x4 __attribute__((ext_vector_type(4)));
typedef u32    u32x2 __attribute__((ext_vector_type(2)));
typedef float  f32x4 __attribute__((ext_vector_type(4)));
typedef unsigned int v4u __attribute__((ext_vector_type(4)));
typedef int    v4i   __attribute__((ext_vector_type(4)));
typedef int    v8i   __attribute__((ext_vector_type(8)));

#define NB   2
#define NT   2048
#define NC   2048
#define NHQ  16
#define NHKV 4
#define NDH  128
#define NWIN 512
#define NQKVD 3072          // (HQ + 2*HKV) * DH
#define NM   (NB * NT)      // 4096 rows

union Frag16 { u32x4 q[2]; v16bf v; };

__device__ __forceinline__ u16 f2bf(float x) {
  u32 u = __float_as_uint(x);
  return (u16)((u + 0x7FFFu + ((u >> 16) & 1u)) >> 16);   // RNE
}
__device__ __forceinline__ float bf2f(u16 h) { return __uint_as_float(((u32)h) << 16); }
__device__ __forceinline__ v8f v8f_zero() {
  v8f r = {0.f, 0.f, 0.f, 0.f, 0.f, 0.f, 0.f, 0.f};
  return r;
}

// XOR-butterfly via ds_swizzle: pattern must be an integer constant expression,
// so the mask is a template parameter. group-of-32 mode: and=0x1F, or=0, xor=M.
template <int M>
__device__ __forceinline__ float xor_swz(float v) {
  return __int_as_float(
      __builtin_amdgcn_ds_swizzle(__float_as_int(v), (M << 10) | 0x1F));
}
__device__ __forceinline__ float half16_max(float v) {   // reduce within 16-lane half
  v = fmaxf(v, xor_swz<1>(v));
  v = fmaxf(v, xor_swz<2>(v));
  v = fmaxf(v, xor_swz<4>(v));
  v = fmaxf(v, xor_swz<8>(v));
  return v;
}
__device__ __forceinline__ float half16_sum(float v) {
  v += xor_swz<1>(v);
  v += xor_swz<2>(v);
  v += xor_swz<4>(v);
  v += xor_swz<8>(v);
  return v;
}
__device__ __forceinline__ float wave32_sum(float v) {
  v += xor_swz<1>(v);
  v += xor_swz<2>(v);
  v += xor_swz<4>(v);
  v += xor_swz<8>(v);
  v += xor_swz<16>(v);
  return v;
}

// ---------------- Tensor Data Mover path (gfx1250 TDM) -------------------
#ifndef USE_TDM
#define USE_TDM 1
#endif
#if USE_TDM
#if __has_builtin(__builtin_amdgcn_tensor_load_to_lds)
#define HAVE_TDM_PATH 1
#endif
#endif

__device__ __forceinline__ void wait_tensor0() {
#if __has_builtin(__builtin_amdgcn_s_wait_tensorcnt)
  __builtin_amdgcn_s_wait_tensorcnt(0);
#else
  asm volatile("s_wait_tensorcnt 0" ::: "memory");
#endif
}

#if defined(HAVE_TDM_PATH)
// 2D tile load: bf16 elements, tile (tile_w x tile_h), row stride `stride` elems.
__device__ __forceinline__ void tdm_load_2d(u32 lds, const void* g, u32 tensor_w,
                                            u32 tensor_h, u32 tile_w, u32 tile_h,
                                            u64 stride_elems) {
  u64 ga = (u64)(size_t)g;
  v4u g0;
  g0.x = 1u;                                             // count=1, user mode
  g0.y = lds;                                            // lds_addr
  g0.z = (u32)(ga & 0xFFFFFFFFu);                        // global_addr[31:0]
  g0.w = (u32)((ga >> 32) & 0x1FFFFFFu) | (2u << 30);    // addr[56:32] | type=2
  u32 w0 = (1u << 16);                                   // data_size=1 (2 bytes)
  u32 w1 = (tensor_w & 0xFFFFu) << 16;                   // tensor_dim0[15:0]
  u32 w2 = ((tensor_w >> 16) & 0xFFFFu) | ((tensor_h & 0xFFFFu) << 16);
  u32 w3 = ((tensor_h >> 16) & 0xFFFFu) | ((tile_w & 0xFFFFu) << 16);
  u32 w4 = (tile_h & 0xFFFFu);                           // tile_dim2 = 0
  u32 w5 = (u32)(stride_elems & 0xFFFFFFFFull);          // dim0 stride lo
  u32 w6 = (u32)((stride_elems >> 32) & 0xFFFFull);      // stride hi | dim1 stride=0
  u32 w7 = 0u;
  v8i g1 = {(int)w0, (int)w1, (int)w2, (int)w3, (int)w4, (int)w5, (int)w6, (int)w7};
  v4i z4 = {0, 0, 0, 0};
#if __clang_major__ >= 23
  v8i z8 = {0, 0, 0, 0, 0, 0, 0, 0};
  __builtin_amdgcn_tensor_load_to_lds(g0, g1, z4, z4, z8, 0);
#else
  __builtin_amdgcn_tensor_load_to_lds(g0, g1, z4, z4, 0);
#endif
}
#endif

// ---------------- f32 -> bf16 conversion (streaming, non-temporal) --------
__global__ __launch_bounds__(256) void cvt_f32_to_bf16(const float* __restrict__ s,
                                                       u16* __restrict__ d, int n) {
  const int i = (blockIdx.x * 256 + threadIdx.x) * 4;
  if (i >= n) return;
  f32x4 f = __builtin_nontemporal_load((const f32x4*)(s + i));
  u32x2 pk;
  pk.x = (u32)f2bf(f.x) | ((u32)f2bf(f.y) << 16);
  pk.y = (u32)f2bf(f.z) | ((u32)f2bf(f.w) << 16);
  __builtin_nontemporal_store(pk, (u32x2*)(d + i));
}

// ---------------- Tiled bf16 WMMA GEMM: C[m][n] = sum_k A[m][k]*B[n][k] ---
// Block tile 128x128, K-step 64, double-buffered TDM pipeline.
// 8 waves: wave (w>>1) -> 32-row strip, (w&1) -> 64-col strip;
// each wave owns a 2x4 grid of 16x16 WMMA tiles (16 WMMA per K-step).
template <int OUTF32>
__global__ __launch_bounds__(256) void gemm_bf16_nt(const u16* __restrict__ A,
                                                    const u16* __restrict__ Bw,
                                                    float* __restrict__ Cf,
                                                    u16* __restrict__ Cb,
                                                    int M, int N, int K) {
  __shared__ alignas(16) u16 As[2][128][64];
  __shared__ alignas(16) u16 Bs[2][128][64];
  const int tid = threadIdx.x;
  const int w = tid >> 5, l = tid & 31, lm = l & 15, hi = (l >> 4) & 1;
  const int m0 = blockIdx.y * 128;
  const int n0 = blockIdx.x * 128;
  const int wm = (w >> 1) * 32;
  const int wn = (w & 1) * 64;

  v8f acc[2][4];
#pragma unroll
  for (int mt = 0; mt < 2; ++mt)
#pragma unroll
    for (int nt = 0; nt < 4; ++nt) acc[mt][nt] = v8f_zero();

  auto compute_step = [&](int buf) {
#pragma unroll
    for (int kc = 0; kc < 2; ++kc) {
      Frag16 fa[2], fb[4];
#pragma unroll
      for (int mt = 0; mt < 2; ++mt) {  // A frag: lane=row, chunks {0..7,16..23}/{8..15,24..31}
        const u16* p = &As[buf][wm + mt * 16 + lm][kc * 32 + hi * 8];
        fa[mt].q[0] = *(const u32x4*)p;
        fa[mt].q[1] = *(const u32x4*)(p + 16);
      }
#pragma unroll
      for (int nt = 0; nt < 4; ++nt) {  // B frag: lane=col, chunks {0..7,16..23}/{8..15,24..31}
        const u16* p = &Bs[buf][wn + nt * 16 + lm][kc * 32 + hi * 8];
        fb[nt].q[0] = *(const u32x4*)p;
        fb[nt].q[1] = *(const u32x4*)(p + 16);
      }
#pragma unroll
      for (int mt = 0; mt < 2; ++mt)
#pragma unroll
        for (int nt = 0; nt < 4; ++nt)
          acc[mt][nt] = __builtin_amdgcn_wmma_f32_16x16x32_bf16(
              false, fa[mt].v, false, fb[nt].v, (short)0, acc[mt][nt], false, false);
    }
  };

#if defined(HAVE_TDM_PATH)
  if (w == 0) {   // prologue: DMA first tiles into buffer 0
    tdm_load_2d((u32)(u64)(size_t)&As[0][0][0], A + (size_t)m0 * K,
                (u32)K, 128u, 64u, 128u, (u64)K);
    tdm_load_2d((u32)(u64)(size_t)&Bs[0][0][0], Bw + (size_t)n0 * K,
                (u32)K, 128u, 64u, 128u, (u64)K);
    wait_tensor0();
  }
  __syncthreads();
  int cur = 0;
  for (int k0 = 0; k0 < K; k0 += 64) {
    const int nxt = cur ^ 1;
    const bool more = (k0 + 64) < K;
    if (w == 0 && more) {   // overlap DMA of next tile with WMMA on current
      tdm_load_2d((u32)(u64)(size_t)&As[nxt][0][0], A + (size_t)m0 * K + k0 + 64,
                  (u32)K, 128u, 64u, 128u, (u64)K);
      tdm_load_2d((u32)(u64)(size_t)&Bs[nxt][0][0], Bw + (size_t)n0 * K + k0 + 64,
                  (u32)K, 128u, 64u, 128u, (u64)K);
    }
    compute_step(cur);
    __syncthreads();                  // everyone done reading `cur`
    if (w == 0 && more) wait_tensor0();
    __syncthreads();                  // `nxt` tile resident in LDS
    cur = nxt;
  }
#else
  for (int k0 = 0; k0 < K; k0 += 64) {
    const int r = tid >> 1;
    const int c = (tid & 1) * 32;
    const u16* ga = A + (size_t)(m0 + r) * K + k0 + c;
    const u16* gb = Bw + (size_t)(n0 + r) * K + k0 + c;
#pragma unroll
    for (int q = 0; q < 4; ++q) {
      *(u32x4*)&As[0][r][c + q * 8] = *(const u32x4*)(ga + q * 8);
      *(u32x4*)&Bs[0][r][c + q * 8] = *(const u32x4*)(gb + q * 8);
    }
    __builtin_prefetch(ga + 64, 0, 1);   // global_prefetch next K tile
    __builtin_prefetch(gb + 64, 0, 1);
    __syncthreads();
    compute_step(0);
    __syncthreads();
  }
#endif

#pragma unroll
  for (int mt = 0; mt < 2; ++mt)
#pragma unroll
    for (int nt = 0; nt < 4; ++nt)
#pragma unroll
      for (int j = 0; j < 8; ++j) {
        const int mr = m0 + wm + mt * 16 + j + hi * 8;   // C/D layout
        const int nc = n0 + wn + nt * 16 + lm;
        if (OUTF32) __builtin_nontemporal_store(acc[mt][nt][j], &Cf[(size_t)mr * N + nc]);
        else        __builtin_nontemporal_store(f2bf(acc[mt][nt][j]), &Cb[(size_t)mr * N + nc]);
      }
}

// ---------------- RoPE + RMSNorm + split/reformat -------------------------
// One wave per (b, t, head-of-24). Q/K: rope+rmsnorm -> [B,H,T,D].
// V: transpose to [B,Hkv,D,T] so attention's P*V B-fragments are contiguous.
__global__ __launch_bounds__(128) void rope_rms_split(const u16* __restrict__ qkv,
                                                      u16* __restrict__ Qo,
                                                      u16* __restrict__ Ko,
                                                      u16* __restrict__ Vt) {
  const int gw = blockIdx.x * 4 + (threadIdx.x >> 5);
  const int l = threadIdx.x & 31;
  const int hh = gw % (NHQ + 2 * NHKV);
  const int row = gw / (NHQ + 2 * NHKV);
  const int t = row % NT;
  const int b = row / NT;
  const u16* src = qkv + (size_t)row * NQKVD + hh * NDH + l * 4;
  const float f0 = bf2f(src[0]), f1 = bf2f(src[1]), f2v = bf2f(src[2]), f3 = bf2f(src[3]);
  if (hh < NHQ + NHKV) {
    // pairs (d0,d0+1),(d0+2,d0+3): freq idx 2l, 2l+1; inv_freq = exp(-ln(1e4)/64 * i)
    const float tf = (float)t;
    float s0, c0, s1, c1;
    __sincosf(tf * __expf(-0.14391156831f * (float)(2 * l)), &s0, &c0);
    __sincosf(tf * __expf(-0.14391156831f * (float)(2 * l + 1)), &s1, &c1);
    const float o0 = f0 * c0 - f1 * s0;
    const float o1 = f1 * c0 + f0 * s0;
    const float o2 = f2v * c1 - f3 * s1;
    const float o3 = f3 * c1 + f2v * s1;
    const float ss = wave32_sum(o0 * o0 + o1 * o1 + o2 * o2 + o3 * o3);
    const float r = rsqrtf(ss * (1.0f / 128.0f) + 1.1920929e-07f);
    u16* dst = (hh < NHQ)
        ? Qo + (((size_t)b * NHQ + hh) * NT + t) * NDH + l * 4
        : Ko + (((size_t)b * NHKV + (hh - NHQ)) * NT + t) * NDH + l * 4;
    u32x2 pk;
    pk.x = (u32)f2bf(o0 * r) | ((u32)f2bf(o1 * r) << 16);
    pk.y = (u32)f2bf(o2 * r) | ((u32)f2bf(o3 * r) << 16);
    *(u32x2*)dst = pk;
  } else {
    const int vh = hh - NHQ - NHKV;
    u16* base = Vt + ((size_t)b * NHKV + vh) * NDH * NT;
    base[(size_t)(l * 4 + 0) * NT + t] = f2bf(f0);
    base[(size_t)(l * 4 + 1) * NT + t] = f2bf(f1);
    base[(size_t)(l * 4 + 2) * NT + t] = f2bf(f2v);
    base[(size_t)(l * 4 + 3) * NT + t] = f2bf(f3);
  }
}

// ---------------- Sliding-window flash attention --------------------------
// Block: (q-tile of 128, b*HQ). 8 waves x 16 q-rows. K-tiles of 32,
// double-buffered TDM pipeline for K and V^T tiles.
__global__ __launch_bounds__(256) void attn_swa(const u16* __restrict__ Qm,
                                                const u16* __restrict__ Km,
                                                const u16* __restrict__ Vtm,
                                                u16* __restrict__ Y) {
  __shared__ alignas(16) u16 Ks[2][32][128];   // [kpos][d]
  __shared__ alignas(16) u16 Vs[2][128][32];   // [d][kpos]
  __shared__ alignas(16) u16 Ps[8][16][32];    // per-wave P tile
  const int tid = threadIdx.x;
  const int w = tid >> 5, l = tid & 31, lm = l & 15, hi = (l >> 4) & 1;
  const int b = blockIdx.y / NHQ, h = blockIdx.y % NHQ, kvh = h >> 2;
  const int q0 = blockIdx.x * 128;
  const int qw = q0 + w * 16;

  // Q fragments: 4 chunks of 32 head-dims, kept in registers
  const u16* qrow = Qm + ((size_t)(b * NHQ + h) * NT + (qw + lm)) * NDH;
  v16bf aq[4];
#pragma unroll
  for (int c = 0; c < 4; ++c) {
    Frag16 f;
    const u16* p = qrow + c * 32 + hi * 8;
    f.q[0] = *(const u32x4*)p;
    f.q[1] = *(const u32x4*)(p + 16);
    aq[c] = f.v;
  }

  v8f o[8];
  float mrun[8], srun[8];
#pragma unroll
  for (int dt = 0; dt < 8; ++dt) o[dt] = v8f_zero();
#pragma unroll
  for (int j = 0; j < 8; ++j) { mrun[j] = -3.0e38f; srun[j] = 0.f; }

  const u16* kbaseptr = Km + (size_t)(b * NHKV + kvh) * NT * NDH;
  const u16* vbaseptr = Vtm + (size_t)(b * NHKV + kvh) * NDH * NT;
  const int ktlo = (q0 + 1 > NWIN) ? ((q0 + 1 - NWIN) >> 5) : 0;
  const int kthi = (q0 + 127) >> 5;

  auto attn_step = [&](int buf, int kb) {
    // S = Q * K^T  (two 16-wide key sub-tiles)
    v8f st[2];
#pragma unroll
    for (int nt = 0; nt < 2; ++nt) {
      v8f s = v8f_zero();
#pragma unroll
      for (int c = 0; c < 4; ++c) {
        Frag16 f;
        const u16* p = &Ks[buf][nt * 16 + lm][c * 32 + hi * 8];
        f.q[0] = *(const u32x4*)p;
        f.q[1] = *(const u32x4*)(p + 16);
        s = __builtin_amdgcn_wmma_f32_16x16x32_bf16(false, aq[c], false, f.v,
                                                    (short)0, s, false, false);
      }
      st[nt] = s;
    }

    // mask + online softmax (row reductions across 16-lane halves)
    const float scale = 0.08838834764831845f;   // 1/sqrt(128)
    float corr[8], p0a[8], p1a[8];
#pragma unroll
    for (int j = 0; j < 8; ++j) {
      const int qi = qw + j + hi * 8;
      const int k0i = kb + lm, k1i = kb + 16 + lm;
      const bool ok0 = (qi >= k0i) && (qi - k0i < NWIN);
      const bool ok1 = (qi >= k1i) && (qi - k1i < NWIN);
      float v0 = ok0 ? st[0][j] * scale : -3.0e38f;
      float v1 = ok1 ? st[1][j] * scale : -3.0e38f;
      const float rm = half16_max(fmaxf(v0, v1));
      const float mnew = fmaxf(mrun[j], rm);
      const float p0 = ok0 ? __expf(v0 - mnew) : 0.f;
      const float p1 = ok1 ? __expf(v1 - mnew) : 0.f;
      const float rs = half16_sum(p0 + p1);
      corr[j] = __expf(mrun[j] - mnew);
      srun[j] = srun[j] * corr[j] + rs;
      mrun[j] = mnew;
      p0a[j] = p0;
      p1a[j] = p1;
    }

    // P -> per-wave LDS (C-layout -> A-layout relayout), rescale O
#pragma unroll
    for (int j = 0; j < 8; ++j) {
      const int r = j + hi * 8;
      Ps[w][r][lm] = f2bf(p0a[j]);
      Ps[w][r][16 + lm] = f2bf(p1a[j]);
    }
#pragma unroll
    for (int dt = 0; dt < 8; ++dt)
#pragma unroll
      for (int j = 0; j < 8; ++j) o[dt][j] *= corr[j];

    Frag16 fp;
    {
      const u16* p = &Ps[w][lm][hi * 8];
      fp.q[0] = *(const u32x4*)p;
      fp.q[1] = *(const u32x4*)(p + 16);
    }
    // O += P * V (8 head-dim tiles)
#pragma unroll
    for (int dt = 0; dt < 8; ++dt) {
      Frag16 fv;
      const u16* p = &Vs[buf][dt * 16 + lm][hi * 8];
      fv.q[0] = *(const u32x4*)p;
      fv.q[1] = *(const u32x4*)(p + 16);
      o[dt] = __builtin_amdgcn_wmma_f32_16x16x32_bf16(false, fp.v, false, fv.v,
                                                      (short)0, o[dt], false, false);
    }
  };

#if defined(HAVE_TDM_PATH)
  auto issue_tiles = [&](int kt, int buf) {
    const int kb = kt * 32;
    tdm_load_2d((u32)(u64)(size_t)&Ks[buf][0][0], kbaseptr + (size_t)kb * NDH,
                (u32)NDH, (u32)NT, (u32)NDH, 32u, (u64)NDH);
    tdm_load_2d((u32)(u64)(size_t)&Vs[buf][0][0], vbaseptr + kb,
                (u32)NT, (u32)NDH, 32u, (u32)NDH, (u64)NT);
  };
  if (w == 0) { issue_tiles(ktlo, 0); wait_tensor0(); }
  __syncthreads();
  int cur = 0;
  for (int kt = ktlo; kt <= kthi; ++kt) {
    const int nxt = cur ^ 1;
    const bool more = kt < kthi;
    if (w == 0 && more) issue_tiles(kt + 1, nxt);   // DMA next tile during compute
    attn_step(cur, kt * 32);
    __syncthreads();
    if (w == 0 && more) wait_tensor0();
    __syncthreads();
    cur = nxt;
  }
#else
  for (int kt = ktlo; kt <= kthi; ++kt) {
    const int kb = kt * 32;
    {   // K tile: contiguous 8KB; V^T tile: 128 rows x 64B
      const u16* src = kbaseptr + (size_t)kb * NDH + tid * 16;
      u16* dst = &Ks[0][0][0] + tid * 16;
      *(u32x4*)dst = *(const u32x4*)src;
      *(u32x4*)(dst + 8) = *(const u32x4*)(src + 8);
      const int r = tid >> 1, c = (tid & 1) * 16;
      const u16* vs = vbaseptr + (size_t)r * NT + kb + c;
      *(u32x4*)&Vs[0][r][c] = *(const u32x4*)vs;
      *(u32x4*)&Vs[0][r][c + 8] = *(const u32x4*)(vs + 8);
    }
    __syncthreads();
    attn_step(0, kb);
    __syncthreads();
  }
#endif

#pragma unroll
  for (int j = 0; j < 8; ++j) srun[j] = (srun[j] > 0.f) ? (1.f / srun[j]) : 0.f;
#pragma unroll
  for (int dt = 0; dt < 8; ++dt)
#pragma unroll
    for (int j = 0; j < 8; ++j) {
      const int qi = qw + j + hi * 8;
      __builtin_nontemporal_store(
          f2bf(o[dt][j] * srun[j]),
          &Y[((size_t)b * NT + qi) * NC + h * NDH + dt * 16 + lm]);
    }
}

// ---------------- launch ---------------------------------------------------
extern "C" void kernel_launch(void* const* d_in, const int* in_sizes, int n_in,
                              void* d_out, int out_size, void* d_ws, size_t ws_size,
                              hipStream_t stream) {
  (void)in_sizes; (void)n_in; (void)out_size; (void)ws_size;
  const float* x = (const float*)d_in[0];
  const float* w_qkv = (const float*)d_in[1];
  const float* w_proj = (const float*)d_in[2];
  float* out = (float*)d_out;

  char* ws = (char*)d_ws;
  size_t off = 0;
  u16* xb     = (u16*)(ws + off); off += (size_t)NM * NC * 2;          // 16 MB
  u16* wqkvb  = (u16*)(ws + off); off += (size_t)NQKVD * NC * 2;       // 12 MB
  u16* wprojb = (u16*)(ws + off); off += (size_t)NC * NC * 2;          // 8 MB
  u16* qkvraw = (u16*)(ws + off); off += (size_t)NM * NQKVD * 2;       // 24 MB
  u16* Qb     = (u16*)(ws + off); off += (size_t)NB * NHQ * NT * NDH * 2;
  u16* Kb     = (u16*)(ws + off); off += (size_t)NB * NHKV * NT * NDH * 2;
  u16* Vtb    = (u16*)(ws + off); off += (size_t)NB * NHKV * NT * NDH * 2;
  u16* Yb     = (u16*)(ws + off); off += (size_t)NM * NC * 2;

  // 1) convert inputs to bf16
  cvt_f32_to_bf16<<<(NM * NC / 4 + 255) / 256, 256, 0, stream>>>(x, xb, NM * NC);
  cvt_f32_to_bf16<<<(NQKVD * NC / 4 + 255) / 256, 256, 0, stream>>>(w_qkv, wqkvb, NQKVD * NC);
  cvt_f32_to_bf16<<<(NC * NC / 4 + 255) / 256, 256, 0, stream>>>(w_proj, wprojb, NC * NC);

  // 2) QKV GEMM (bf16 out)
  dim3 g1(NQKVD / 128, NM / 128);
  gemm_bf16_nt<0><<<g1, 256, 0, stream>>>(xb, wqkvb, (float*)nullptr, qkvraw,
                                          NM, NQKVD, NC);

  // 3) RoPE + RMSNorm + split (V transposed)
  rope_rms_split<<<(NM * (NHQ + 2 * NHKV)) / 4, 128, 0, stream>>>(qkvraw, Qb, Kb, Vtb);

  // 4) sliding-window flash attention
  dim3 g2(NT / 128, NB * NHQ);
  attn_swa<<<g2, 256, 0, stream>>>(Qb, Kb, Vtb, Yb);

  // 5) output projection (f32 out)
  dim3 g3(NC / 128, NM / 128);
  gemm_bf16_nt<1><<<g3, 256, 0, stream>>>(Yb, wprojb, out, (u16*)nullptr, NM, NC, NC);
}
